// TransformerV1_19559281066518
// MI455X (gfx1250) — matrix operand order repaired and checked by reading, not compiled, not run on hardware
//
#include <hip/hip_runtime.h>
#include <hip/hip_bf16.h>
#include <cmath>

// ---------------------------------------------------------------------------
// Types for CDNA5 WMMA
// ---------------------------------------------------------------------------
typedef __attribute__((ext_vector_type(16))) __bf16       v16bf;
typedef __attribute__((ext_vector_type(8)))  float        v8f;
typedef __attribute__((ext_vector_type(4)))  unsigned int v4u;

union FragU { v4u u[2]; v16bf v; };

// fp32 -> bf16 (round-half-up): 2 VALU
__device__ __forceinline__ unsigned short f2bf(float f) {
  return (unsigned short)((__float_as_uint(f) + 0x8000u) >> 16);
}

// two fp32 -> packed bf16x2 in one u32 (lo = a, hi = b):
// one v_add per value + one v_perm_b32 byte-splice -> 3 VALU per pair
__device__ __forceinline__ unsigned pack2bf(float a, float b) {
  unsigned ua = __float_as_uint(a) + 0x8000u;
  unsigned ub = __float_as_uint(b) + 0x8000u;
  // result bytes (LSB..MSB) = ua.b2, ua.b3, ub.b2, ub.b3
  return __builtin_amdgcn_perm(ub, ua, 0x07060302u);
}

#define TM 128
#define TN 128
#define TK 32
#define LDA_S (TK + 8)    // 40 shorts / row
#define LDB_S (TN + 8)    // 136 shorts / row

// ---------------------------------------------------------------------------
// WMMA GEMM: C[M,N] = act( alpha * A[M,K] @ B + bias ), B = [K,N] (NN) or
// [N,K] (NT). Batched via blockIdx.z with element strides sA/sB/sC.
//
// Block = 256 threads = 8 wave32 arranged 4(M) x 2(N). Each wave owns a
// 32x64 register tile = 2x4 WMMA tiles -> 8 v_wmma_f32_16x16x32_bf16 per
// K-step. LDS is double-buffered and global loads for step t+1 are register-
// staged during step t's compute; one barrier per K-step.
// Per step: 32 KB fp32 in, 128*128*32*2 = 1.05 MFLOP -> ~32 FLOP/byte.
// Dims are multiples of (128, 128, 32) for every GEMM in this model.
// ---------------------------------------------------------------------------
__global__ __launch_bounds__(256)
void wmma_gemm_kernel(const float* __restrict__ A, const float* __restrict__ Bm,
                      const float* __restrict__ bias, float* __restrict__ C,
                      int M, int N, int K, int transB, int relu, float alpha,
                      long sA, long sB, long sC)
{
  A  += (size_t)blockIdx.z * sA;
  Bm += (size_t)blockIdx.z * sB;
  C  += (size_t)blockIdx.z * sC;

  const int blockM = blockIdx.y * TM;
  const int blockN = blockIdx.x * TN;
  const int tid  = threadIdx.x;
  const int lane = tid & 31;
  const int wave = tid >> 5;
  const int wm   = wave & 3;        // 4 wave tiles in M
  const int wn   = wave >> 2;       // 2 wave tiles in N
  const int lh   = lane >> 4;       // lane half (0/1)
  const int lr   = lane & 15;
  const int ldb  = transB ? K : N;

  // staging index maps (same for load & store phases)
  const int ar = tid >> 1;          // A row 0..127
  const int ac = (tid & 1) * 16;    // A col 0 or 16
  const int br = tid >> 3;          // B row 0..31   (NN)
  const int bc = (tid & 7) * 16;    // B col 0..112  (NN)
  const int bn = tid >> 1;          // B n  0..127   (NT)
  const int bk = (tid & 1) * 16;    // B k  0 or 16  (NT)

  __shared__ __align__(16) unsigned short As[2][TM][LDA_S]; // 128x32 bf16 x2
  __shared__ __align__(16) unsigned short Bs[2][TK][LDB_S]; // 32x128 bf16 x2

  v8f acc[2][4];
#pragma unroll
  for (int i = 0; i < 2; ++i)
#pragma unroll
    for (int j = 0; j < 4; ++j)
      acc[i][j] = (v8f){0.f,0.f,0.f,0.f,0.f,0.f,0.f,0.f};

  float4 fa0, fa1, fa2, fa3, fb0, fb1, fb2, fb3;

  // ---- register-stage one K-step of A and B from global ----
  auto load_step = [&](int k0) {
    const float4* pa =
        reinterpret_cast<const float4*>(A + (size_t)(blockM + ar) * K + k0 + ac);
    fa0 = pa[0]; fa1 = pa[1]; fa2 = pa[2]; fa3 = pa[3];
    if (!transB) {
      const float4* pb =
          reinterpret_cast<const float4*>(Bm + (size_t)(k0 + br) * ldb + blockN + bc);
      fb0 = pb[0]; fb1 = pb[1]; fb2 = pb[2]; fb3 = pb[3];
    } else {
      const float4* pb =
          reinterpret_cast<const float4*>(Bm + (size_t)(blockN + bn) * ldb + k0 + bk);
      fb0 = pb[0]; fb1 = pb[1]; fb2 = pb[2]; fb3 = pb[3];
    }
  };

  // ---- convert staged registers to bf16 and store into LDS buffer `buf` ----
  auto store_step = [&](int buf) {
    v4u pkA0 = { pack2bf(fa0.x, fa0.y), pack2bf(fa0.z, fa0.w),
                 pack2bf(fa1.x, fa1.y), pack2bf(fa1.z, fa1.w) };
    v4u pkA1 = { pack2bf(fa2.x, fa2.y), pack2bf(fa2.z, fa2.w),
                 pack2bf(fa3.x, fa3.y), pack2bf(fa3.z, fa3.w) };
    *reinterpret_cast<v4u*>(&As[buf][ar][ac])     = pkA0;
    *reinterpret_cast<v4u*>(&As[buf][ar][ac + 8]) = pkA1;
    if (!transB) {
      v4u pkB0 = { pack2bf(fb0.x, fb0.y), pack2bf(fb0.z, fb0.w),
                   pack2bf(fb1.x, fb1.y), pack2bf(fb1.z, fb1.w) };
      v4u pkB1 = { pack2bf(fb2.x, fb2.y), pack2bf(fb2.z, fb2.w),
                   pack2bf(fb3.x, fb3.y), pack2bf(fb3.z, fb3.w) };
      *reinterpret_cast<v4u*>(&Bs[buf][br][bc])     = pkB0;
      *reinterpret_cast<v4u*>(&Bs[buf][br][bc + 8]) = pkB1;
    } else {
      // scatter-transpose: B[N,K] row -> LDS column
      Bs[buf][bk +  0][bn] = f2bf(fb0.x); Bs[buf][bk +  1][bn] = f2bf(fb0.y);
      Bs[buf][bk +  2][bn] = f2bf(fb0.z); Bs[buf][bk +  3][bn] = f2bf(fb0.w);
      Bs[buf][bk +  4][bn] = f2bf(fb1.x); Bs[buf][bk +  5][bn] = f2bf(fb1.y);
      Bs[buf][bk +  6][bn] = f2bf(fb1.z); Bs[buf][bk +  7][bn] = f2bf(fb1.w);
      Bs[buf][bk +  8][bn] = f2bf(fb2.x); Bs[buf][bk +  9][bn] = f2bf(fb2.y);
      Bs[buf][bk + 10][bn] = f2bf(fb2.z); Bs[buf][bk + 11][bn] = f2bf(fb2.w);
      Bs[buf][bk + 12][bn] = f2bf(fb3.x); Bs[buf][bk + 13][bn] = f2bf(fb3.y);
      Bs[buf][bk + 14][bn] = f2bf(fb3.z); Bs[buf][bk + 15][bn] = f2bf(fb3.w);
    }
  };

  const int nsteps = K / TK;

  // prologue: fill buffer 0
  load_step(0);
  store_step(0);
  __syncthreads();

  for (int t = 0; t < nsteps; ++t) {
    const int cur = t & 1;
    const bool more = (t + 1) < nsteps;

    if (more) load_step((t + 1) * TK);   // global loads overlap this step's math
    if (t + 2 < nsteps)                  // L2 lookahead (global_prefetch_b8)
      __builtin_prefetch(A + (size_t)(blockM + ar) * K + (t + 2) * TK + ac, 0, 1);

    // ---- fragment loads per ISA VGPR layouts ----
    // A 16x32 bf16: lane half 0 -> K [0..7],[16..23]; half 1 -> K [8..15],[24..31]
    FragU a[2], b[4];
    const int ra = wm * 32 + lr;
#pragma unroll
    for (int i = 0; i < 2; ++i) {
      a[i].u[0] = *reinterpret_cast<const v4u*>(&As[cur][ra + 16 * i][8 * lh]);
      a[i].u[1] = *reinterpret_cast<const v4u*>(&As[cur][ra + 16 * i][16 + 8 * lh]);
    }
    // B 32x16 bf16: lane = N column; half 0 -> K [0..7],[16..23]; half 1 -> K [8..15],[24..31]
    const int cb = wn * 64;
#pragma unroll
    for (int j = 0; j < 4; ++j) {
      const int nb = cb + 16 * j + lr;
#pragma unroll
      for (int q = 0; q < 4; ++q) {
        b[j].u[0][q] = (unsigned)Bs[cur][8 * lh + 2 * q][nb] | ((unsigned)Bs[cur][8 * lh + 2 * q + 1][nb] << 16);
        b[j].u[1][q] = (unsigned)Bs[cur][16 + 8 * lh + 2 * q][nb] | ((unsigned)Bs[cur][16 + 8 * lh + 2 * q + 1][nb] << 16);
      }
    }
#pragma unroll
    for (int i = 0; i < 2; ++i)
#pragma unroll
      for (int j = 0; j < 4; ++j)
        acc[i][j] = __builtin_amdgcn_wmma_f32_16x16x32_bf16(
            false, a[i].v, false, b[j].v, (short)0, acc[i][j], false, false);

    if (more) {
      store_step(1 - cur);   // readers of that buffer finished before last barrier
      __syncthreads();
    }
  }

  // ---- epilogue: C/D layout lane 0-15: N=lane, M=r; lane 16-31: N=lane-16, M=r+8
  const int gnBase = blockN + wn * 64 + lr;
  const int gmBase = blockM + wm * 32 + 8 * lh;
  float bv[4];
#pragma unroll
  for (int j = 0; j < 4; ++j) bv[j] = bias ? bias[gnBase + 16 * j] : 0.f;
#pragma unroll
  for (int i = 0; i < 2; ++i) {
#pragma unroll
    for (int r = 0; r < 8; ++r) {
      const int gm = gmBase + 16 * i + r;
#pragma unroll
      for (int j = 0; j < 4; ++j) {
        float vv = acc[i][j][r] * alpha + bv[j];
        if (relu) vv = fmaxf(vv, 0.f);
        C[(size_t)gm * N + gnBase + 16 * j] = vv;
      }
    }
  }
}

// ---------------------------------------------------------------------------
// Embedding gather: out[m,:] = embed[seq[m],:]
// ---------------------------------------------------------------------------
__global__ __launch_bounds__(256)
void gather_kernel(const int* __restrict__ seq, const float* __restrict__ embed,
                   float* __restrict__ out, int E)
{
  const int m = blockIdx.x;
  const int v = seq[m];
  for (int e = threadIdx.x; e < E; e += blockDim.x)
    out[(size_t)m * E + e] = embed[(size_t)v * E + e];
}

// ---------------------------------------------------------------------------
// RoPE in place, one block per token row, one thread per (even,odd) pair.
// ---------------------------------------------------------------------------
__global__ void rope_kernel(float* __restrict__ x, int S, int E)
{
  const int m = blockIdx.x;
  const int s = m % S;
  const int i = threadIdx.x;              // 0 .. E/2-1
  const float inv_freq = __expf(-(2.f * i / (float)E) * 9.210340371976184f); // ln(1e4)
  float sn, cs;
  __sincosf((float)s * inv_freq, &sn, &cs);
  float* p = x + (size_t)m * E + 2 * i;
  const float x1 = p[0], x2 = p[1];
  p[0] = x1 * cs - x2 * sn;
  p[1] = x1 * sn + x2 * cs;
}

// ---------------------------------------------------------------------------
// Masked softmax over score rows: scores[b,s,:], pad mask + optional causal.
// ---------------------------------------------------------------------------
__global__ __launch_bounds__(256)
void softmax_kernel(float* __restrict__ scores, const unsigned char* __restrict__ pad,
                    int S, int causal)
{
  const int s = blockIdx.x, b = blockIdx.y;
  float* row = scores + ((size_t)b * S + s) * S;
  const unsigned char* pm = pad + (size_t)b * S;
  const int tid = threadIdx.x;
  __shared__ float red[256];

  float mx = -3.402823466e38f;
  for (int t = tid; t < S; t += 256) {
    bool masked = pm[t] || (causal && t > s);
    if (!masked) mx = fmaxf(mx, row[t]);
  }
  red[tid] = mx; __syncthreads();
  for (int o = 128; o > 0; o >>= 1) {
    if (tid < o) red[tid] = fmaxf(red[tid], red[tid + o]);
    __syncthreads();
  }
  mx = red[0]; __syncthreads();

  float sum = 0.f;
  for (int t = tid; t < S; t += 256) {
    bool masked = pm[t] || (causal && t > s);
    if (!masked) sum += __expf(row[t] - mx);
  }
  red[tid] = sum; __syncthreads();
  for (int o = 128; o > 0; o >>= 1) {
    if (tid < o) red[tid] += red[tid + o];
    __syncthreads();
  }
  sum = red[0]; __syncthreads();
  const float inv = (sum > 0.f) ? 1.f / sum : 0.f;

  for (int t = tid; t < S; t += 256) {
    bool masked = pm[t] || (causal && t > s);
    row[t] = masked ? 0.f : __expf(row[t] - mx) * inv;
  }
}

// ---------------------------------------------------------------------------
// out = LayerNorm(x + h) * gamma + beta  (safe when out == x)
// ---------------------------------------------------------------------------
__global__ __launch_bounds__(256)
void add_ln_kernel(const float* xin, const float* __restrict__ h,
                   const float* __restrict__ gamma, const float* __restrict__ beta,
                   float* out, int E)
{
  const int m = blockIdx.x;
  const int tid = threadIdx.x;
  __shared__ float r1[256], r2[256];
  float sum = 0.f, sq = 0.f;
  for (int e = tid; e < E; e += 256) {
    float v = xin[(size_t)m * E + e] + h[(size_t)m * E + e];
    sum += v; sq += v * v;
  }
  r1[tid] = sum; r2[tid] = sq; __syncthreads();
  for (int o = 128; o > 0; o >>= 1) {
    if (tid < o) { r1[tid] += r1[tid + o]; r2[tid] += r2[tid + o]; }
    __syncthreads();
  }
  const float mu   = r1[0] / (float)E;
  const float var  = r2[0] / (float)E - mu * mu;
  const float rstd = rsqrtf(var + 1e-5f);
  for (int e = tid; e < E; e += 256) {
    float v = xin[(size_t)m * E + e] + h[(size_t)m * E + e];
    out[(size_t)m * E + e] = (v - mu) * rstd * gamma[e] + beta[e];
  }
}

// ---------------------------------------------------------------------------
// Host orchestration
// ---------------------------------------------------------------------------
extern "C" void kernel_launch(void* const* d_in, const int* in_sizes, int n_in,
                              void* d_out, int out_size, void* d_ws, size_t ws_size,
                              hipStream_t stream)
{
  (void)in_sizes; (void)n_in; (void)out_size; (void)ws_size;
  const int B = 4, S = 1024, E = 768, H = 3072, L = 5, V = 16384;
  const int M = B * S;

  const int*           seq        = (const int*)d_in[0];
  const unsigned char* pad        = (const unsigned char*)d_in[1];
  const float* embed      = (const float*)d_in[2];
  const float* enc_qkv_w  = (const float*)d_in[3];
  const float* enc_qkv_b  = (const float*)d_in[4];
  const float* enc_ln     = (const float*)d_in[5];
  const float* enc_mlp_w1 = (const float*)d_in[6];
  const float* enc_mlp_b1 = (const float*)d_in[7];
  const float* enc_mlp_w2 = (const float*)d_in[8];
  const float* enc_mlp_b2 = (const float*)d_in[9];
  const float* dec_qkv_w  = (const float*)d_in[10];
  const float* dec_qkv_b  = (const float*)d_in[11];
  const float* dec_ln     = (const float*)d_in[12];
  const float* dec_mlp_w1 = (const float*)d_in[13];
  const float* dec_mlp_b1 = (const float*)d_in[14];
  const float* dec_mlp_w2 = (const float*)d_in[15];
  const float* dec_mlp_b2 = (const float*)d_in[16];
  const float* unembed    = (const float*)d_in[17];

  float* ws = (float*)d_ws;
  size_t off = 0;
  auto alloc = [&](size_t n) { float* p = ws + off; off += n; return p; };
  float* x        = alloc((size_t)M * E);
  float* embeds   = alloc((size_t)M * E);
  float* q        = alloc((size_t)M * E);
  float* kbuf     = alloc((size_t)M * E);
  float* vbuf     = alloc((size_t)M * E);
  float* h        = alloc((size_t)M * E);
  float* hid      = alloc((size_t)M * H);
  float* scores   = alloc((size_t)B * S * S);
  float* enc_outs = alloc((size_t)L * M * E);

  const float scal = 1.0f / sqrtf((float)E);
  const size_t rowBytes = (size_t)M * E * sizeof(float);

  auto gemm = [&](const float* Ap, const float* Bp, const float* biasp, float* Cp,
                  int Mv, int Nv, int Kv, int transB, int relu, float alpha,
                  long sA, long sB, long sC, int batch) {
    dim3 grid(Nv / TN, Mv / TM, batch);
    wmma_gemm_kernel<<<grid, dim3(256), 0, stream>>>(Ap, Bp, biasp, Cp, Mv, Nv, Kv,
                                                     transB, relu, alpha, sA, sB, sC);
  };
  auto rope = [&](float* p) { rope_kernel<<<dim3(M), dim3(E / 2), 0, stream>>>(p, S, E); };
  auto addln = [&](const float* xi, const float* hi, const float* ln_base, float* o) {
    add_ln_kernel<<<dim3(M), dim3(256), 0, stream>>>(xi, hi, ln_base, ln_base + E, o, E);
  };
  auto attn = [&](float* hq, float* hk, float* hv, float* hout, int causal) {
    rope(hq); rope(hk);
    gemm(hq, hk, nullptr, scores, S, S, E, 1, 0, scal,
         (long)S * E, (long)S * E, (long)S * S, B);
    softmax_kernel<<<dim3(S, B), dim3(256), 0, stream>>>(scores, pad, S, causal);
    gemm(scores, hv, nullptr, hout, S, E, S, 0, 0, 1.f,
         (long)S * S, (long)S * E, (long)S * E, B);
  };

  gather_kernel<<<dim3(M), dim3(256), 0, stream>>>(seq, embed, embeds, E);
  hipMemcpyAsync(x, embeds, rowBytes, hipMemcpyDeviceToDevice, stream);

  // ---------------- encoder ----------------
  for (int i = 0; i < L; ++i) {
    const float* W  = enc_qkv_w + (size_t)i * 3 * E * E;
    const float* bb = enc_qkv_b + (size_t)i * 3 * E;
    gemm(x, W + 0 * (size_t)E * E, bb + 0 * E, q,    M, E, E, 0, 0, 1.f, 0, 0, 0, 1);
    gemm(x, W + 1 * (size_t)E * E, bb + 1 * E, kbuf, M, E, E, 0, 0, 1.f, 0, 0, 0, 1);
    gemm(x, W + 2 * (size_t)E * E, bb + 2 * E, vbuf, M, E, E, 0, 0, 1.f, 0, 0, 0, 1);
    attn(q, kbuf, vbuf, h, /*causal=*/0);
    addln(x, h, enc_ln + ((size_t)i * 2 + 0) * 2 * E, x);
    gemm(x,   enc_mlp_w1 + (size_t)i * E * H, enc_mlp_b1 + (size_t)i * H, hid, M, H, E, 0, 1, 1.f, 0, 0, 0, 1);
    gemm(hid, enc_mlp_w2 + (size_t)i * H * E, enc_mlp_b2 + (size_t)i * E, h,   M, E, H, 0, 1, 1.f, 0, 0, 0, 1);
    addln(x, h, enc_ln + ((size_t)i * 2 + 1) * 2 * E, x);
    hipMemcpyAsync(enc_outs + (size_t)i * M * E, x, rowBytes, hipMemcpyDeviceToDevice, stream);
  }

  // ---------------- decoder ----------------
  hipMemcpyAsync(x, embeds, rowBytes, hipMemcpyDeviceToDevice, stream);
  for (int i = 0; i < L; ++i) {
    const float* W  = dec_qkv_w + (size_t)i * 6 * E * E;
    const float* bb = dec_qkv_b + (size_t)i * 6 * E;
    const float* kv = enc_outs + (size_t)i * M * E;
    // self-attention (causal)
    gemm(x, W + 0 * (size_t)E * E, bb + 0 * E, q,    M, E, E, 0, 0, 1.f, 0, 0, 0, 1);
    gemm(x, W + 1 * (size_t)E * E, bb + 1 * E, kbuf, M, E, E, 0, 0, 1.f, 0, 0, 0, 1);
    gemm(x, W + 2 * (size_t)E * E, bb + 2 * E, vbuf, M, E, E, 0, 0, 1.f, 0, 0, 0, 1);
    attn(q, kbuf, vbuf, h, /*causal=*/1);
    addln(x, h, dec_ln + ((size_t)i * 3 + 0) * 2 * E, x);
    // cross-attention over encoder output
    gemm(x,  W + 3 * (size_t)E * E, bb + 3 * E, q,    M, E, E, 0, 0, 1.f, 0, 0, 0, 1);
    gemm(kv, W + 4 * (size_t)E * E, bb + 4 * E, kbuf, M, E, E, 0, 0, 1.f, 0, 0, 0, 1);
    gemm(kv, W + 5 * (size_t)E * E, bb + 5 * E, vbuf, M, E, E, 0, 0, 1.f, 0, 0, 0, 1);
    attn(q, kbuf, vbuf, h, /*causal=*/0);
    addln(x, h, dec_ln + ((size_t)i * 3 + 1) * 2 * E, x);
    // MLP
    gemm(x,   dec_mlp_w1 + (size_t)i * E * H, dec_mlp_b1 + (size_t)i * H, hid, M, H, E, 0, 1, 1.f, 0, 0, 0, 1);
    gemm(hid, dec_mlp_w2 + (size_t)i * H * E, dec_mlp_b2 + (size_t)i * E, h,   M, E, H, 0, 1, 1.f, 0, 0, 0, 1);
    addln(x, h, dec_ln + ((size_t)i * 3 + 2) * 2 * E, x);
  }

  // ---------------- unembed ----------------
  gemm(x, unembed, nullptr, (float*)d_out, M, V, E, 0, 0, 1.f, 0, 0, 0, 1);
}
